// MultiAgentDecisionTransformer_84662395338819
// MI455X (gfx1250) — matrix operand order repaired and checked by reading, not compiled, not run on hardware
//
#include <hip/hip_runtime.h>
#include <hip/hip_bf16.h>

// ---------------- problem dims ----------------
static constexpr int Bn   = 32;
static constexpr int Nn   = 128;
static constexpr int Tt   = 20;
static constexpr int SD   = 64;
static constexpr int AD   = 8;
static constexpr int H    = 256;
static constexpr int NH   = 8;
static constexpr int DH   = 32;   // H/NH
static constexpr int GH   = 4;
static constexpr int GL   = 2;
static constexpr int NL   = 3;
static constexpr int S3   = 60;   // 3*T
static constexpr int TOK  = Bn * Nn * Tt;    // 81920
static constexpr int SEQR = Bn * Nn * S3;    // 245760
static constexpr int CH_SAMP = 1024;         // samples per chunk
static constexpr int CH_ROWS = CH_SAMP * S3; // 61440
static constexpr float EPSF = 1e-5f;

typedef __attribute__((ext_vector_type(16))) __bf16 v16bf;
typedef __attribute__((ext_vector_type(8)))  __bf16 v8bf;
typedef __attribute__((ext_vector_type(8)))  float  v8f;

// ---------------- helpers ----------------
__device__ inline float bf2f(__bf16 b) {
  unsigned u = ((unsigned)__builtin_bit_cast(unsigned short, b)) << 16;
  return __builtin_bit_cast(float, u);
}
__device__ inline __bf16 f2bf(float f) {
  unsigned u = __builtin_bit_cast(unsigned, f);
  u += 0x7FFFu + ((u >> 16) & 1u);               // round-to-nearest-even
  unsigned short h = (unsigned short)(u >> 16);
  return __builtin_bit_cast(__bf16, h);
}
__device__ inline unsigned pack2(float a, float b) {
  unsigned lo = (unsigned)__builtin_bit_cast(unsigned short, f2bf(a));
  unsigned hi = (unsigned)__builtin_bit_cast(unsigned short, f2bf(b));
  return lo | (hi << 16);
}
__device__ inline float toF(float v)  { return v; }
__device__ inline float toF(__bf16 v) { return bf2f(v); }
__device__ inline void stC(float* p,  float v) { *p = v; }
__device__ inline void stC(__bf16* p, float v) { *p = f2bf(v); }

__device__ inline float geluf(float x) {
  return 0.5f * x * (1.0f + erff(x * 0.70710678118654752f));
}
__device__ inline float wsum(float v) {
#pragma unroll
  for (int o = 16; o > 0; o >>= 1) v += __shfl_xor(v, o, 32);
  return v;
}
__device__ inline float wmaxr(float v) {
#pragma unroll
  for (int o = 16; o > 0; o >>= 1) v = fmaxf(v, __shfl_xor(v, o, 32));
  return v;
}

// ---------------- WMMA GEMM ----------------
// C[M,N] (+)= epi( A[M,K] * B[K,N] + bias )
// REQUIRES: M % 128 == 0 (or grid sized exactly), N % 64 == 0, K % 32 == 0.
// All launches in this network satisfy this, so there is no guarded path.
// A: row-major f32 or bf16; B: row-major f32 (converted to bf16, transposed
// into LDS).  f32 accumulate via v_wmma_f32_16x16x32_bf16.
// EPI: 0=none, 1=+bias, 2=+bias then exact GELU.  ACCUM: C += result.
static constexpr int BM = 128, BN = 64, BK = 32;
static constexpr int ASTR  = 40;   // u16 per A row (32 data + 8 pad) -> 80B, 16B-aligned
static constexpr int BTSTR = 40;   // u16 per B column (transposed tile)

template <typename AT, typename CT, int EPI, bool ACCUM>
__global__ __launch_bounds__(256) void gemm_wmma(
    const AT* __restrict__ A, long long sAb, int lda,
    const float* __restrict__ Bw, long long sBb, int ldb,
    CT* __restrict__ C, long long sCb, int ldc,
    int K, const float* __restrict__ bias)
{
  __shared__ __bf16 As[BM * ASTR];    // [row][k]
  __shared__ __bf16 Bt[BN * BTSTR];   // [col][k]  (transposed)
  const int tid  = threadIdx.x;
  const int wv   = tid >> 5;
  const int lane = tid & 31;
  const int hf   = lane >> 4;
  const int l16  = lane & 15;

  const AT*    Ab = A  + (long long)blockIdx.z * sAb;
  const float* Bb = Bw + (long long)blockIdx.z * sBb;
  CT*          Cb = C  + (long long)blockIdx.z * sCb;
  const int rowBase = blockIdx.y * BM;
  const int colBase = blockIdx.x * BN;

  v8f acc[4];
#pragma unroll
  for (int t = 0; t < 4; ++t)
#pragma unroll
    for (int i = 0; i < 8; ++i) acc[t][i] = 0.0f;

  // per-thread staging coordinates (fixed across K loop)
  const int ar = tid >> 1;            // A row 0..127
  const int ac = (tid & 1) * 16;      // A col group 0/16
  const int bkk = tid & 15;           // B k-pair 0..15
  const int bc  = (tid >> 4) * 4;     // B col group (4 cols)

  for (int k0 = 0; k0 < K; k0 += BK) {
    { // A tile: 16 contiguous k's per thread; vector ld/st
      __bf16* dst = &As[ar * ASTR + ac];
      const AT* gp = Ab + (long long)(rowBase + ar) * lda + (k0 + ac);
      if constexpr (sizeof(AT) == 2) {
        const uint4* g4 = (const uint4*)gp;
        ((uint4*)dst)[0] = g4[0];
        ((uint4*)dst)[1] = g4[1];
      } else {
        const float4* g4 = (const float4*)gp;
        float4 f0 = g4[0], f1 = g4[1], f2 = g4[2], f3 = g4[3];
        uint4 u0, u1;
        u0.x = pack2(f0.x, f0.y); u0.y = pack2(f0.z, f0.w);
        u0.z = pack2(f1.x, f1.y); u0.w = pack2(f1.z, f1.w);
        u1.x = pack2(f2.x, f2.y); u1.y = pack2(f2.z, f2.w);
        u1.z = pack2(f3.x, f3.y); u1.w = pack2(f3.z, f3.w);
        ((uint4*)dst)[0] = u0;
        ((uint4*)dst)[1] = u1;
      }
    }
    { // B tile: transpose via u32 pair packs
      const float* g0 = Bb + (long long)(k0 + 2 * bkk) * ldb + (colBase + bc);
      float4 r0 = *(const float4*)g0;
      float4 r1 = *(const float4*)(g0 + ldb);
      unsigned* bt32 = (unsigned*)Bt;
      bt32[(bc + 0) * (BTSTR / 2) + bkk] = pack2(r0.x, r1.x);
      bt32[(bc + 1) * (BTSTR / 2) + bkk] = pack2(r0.y, r1.y);
      bt32[(bc + 2) * (BTSTR / 2) + bkk] = pack2(r0.z, r1.z);
      bt32[(bc + 3) * (BTSTR / 2) + bkk] = pack2(r0.w, r1.w);
    }
    __syncthreads();

    // A fragment: two aligned b128 LDS loads (ISA 16-bit 16x32 layout)
    const int am = wv * 16 + l16;
    v8bf a0 = *(const v8bf*)&As[am * ASTR + hf * 8];
    v8bf a1 = *(const v8bf*)&As[am * ASTR + 16 + hf * 8];
    v16bf afr = __builtin_shufflevector(a0, a1,
        0, 1, 2, 3, 4, 5, 6, 7, 8, 9, 10, 11, 12, 13, 14, 15);

#pragma unroll
    for (int tb = 0; tb < 4; ++tb) {
      const int bnc = tb * 16 + l16;
      v8bf b0 = *(const v8bf*)&Bt[bnc * BTSTR + hf * 8];
      v8bf b1 = *(const v8bf*)&Bt[bnc * BTSTR + 16 + hf * 8];
      v16bf bfr = __builtin_shufflevector(b0, b1,
          0, 1, 2, 3, 4, 5, 6, 7, 8, 9, 10, 11, 12, 13, 14, 15);
      acc[tb] = __builtin_amdgcn_wmma_f32_16x16x32_bf16(
          false, afr, false, bfr, (short)0, acc[tb], false, false);
    }
    __syncthreads();
  }

#pragma unroll
  for (int tb = 0; tb < 4; ++tb) {
    const int gn = colBase + tb * 16 + l16;
#pragma unroll
    for (int i = 0; i < 8; ++i) {
      const int gm = rowBase + wv * 16 + hf * 8 + i;
      float v = acc[tb][i];
      if (EPI >= 1) v += bias[gn];
      if (EPI == 2) v = geluf(v);
      long long off = (long long)gm * ldc + gn;
      if (ACCUM) v += toF(Cb[off]);
      stC(&Cb[off], v);
    }
  }
}

// ---------------- elementwise / wave-per-row kernels ----------------
// finish state embedding: sg[(b*T+t)*N+n] = tmp[(b*N+n)*T+t] + bias + time + agent
__global__ void k_embed_state_finish(const float* __restrict__ tmp,
    const float* __restrict__ bs, const float* __restrict__ Et,
    const float* __restrict__ Eag, const int* __restrict__ tsteps,
    float* __restrict__ sg)
{
  int idx = blockIdx.x * 256 + threadIdx.x;          // TOK*H total
  int h = idx & (H - 1); int r = idx >> 8;
  int t = r % Tt; int bn = r / Tt; int n = bn % Nn; int b = bn / Nn;
  int ts = tsteps[r];
  float v = tmp[idx] + bs[h] + Et[ts * H + h] + Eag[n * H + h];
  sg[((long long)(b * Tt + t) * Nn + n) * H + h] = v;
}

// return & action embeddings straight into seq rows 3t and 3t+2
__global__ void k_embed_ret_act(const float* __restrict__ rtg,
    const int* __restrict__ actions, const int* __restrict__ tsteps,
    const float* __restrict__ Wr, const float* __restrict__ br,
    const float* __restrict__ Eact, const float* __restrict__ Et,
    const float* __restrict__ Eag, float* __restrict__ seq)
{
  int idx = blockIdx.x * 256 + threadIdx.x;          // TOK*H total
  int h = idx & (H - 1); int r = idx >> 8;
  int t = r % Tt; int bn = r / Tt; int n = bn % Nn;
  float te = Et[tsteps[r] * H + h] + Eag[n * H + h];
  long long base = ((long long)bn * S3 + 3 * t) * H + h;
  seq[base]         = rtg[r] * Wr[h] + br[h] + te;
  seq[base + 2 * H] = Eact[actions[r] * H + h] + te;
}

__global__ void k_fill_bias(float* __restrict__ dst, const float* __restrict__ bias) {
  int idx = blockIdx.x * 256 + threadIdx.x;
  dst[idx] = bias[idx & (H - 1)];
}

__global__ void k_row_dots(const float* __restrict__ wh,
    const float* __restrict__ asrc, const float* __restrict__ adst,
    float* __restrict__ es, float* __restrict__ ed, int rows)
{
  int wid = blockIdx.x * 8 + (threadIdx.x >> 5);
  if (wid >= rows) return;
  int lane = threadIdx.x & 31;
  float s1 = 0.f, s2 = 0.f;
#pragma unroll
  for (int i = 0; i < 8; ++i) {
    int h = lane + 32 * i;
    float v = wh[(long long)wid * H + h];
    s1 += v * asrc[h]; s2 += v * adst[h];
  }
  s1 = wsum(s1); s2 = wsum(s2);
  if (lane == 0) { es[wid] = s1; ed[wid] = s2; }
}

__global__ void k_gat_softmax(const float* __restrict__ es,
    const float* __restrict__ ed, const float* __restrict__ adj,
    float* __restrict__ attn, int rowsTotal)
{
  int wid = blockIdx.x * 8 + (threadIdx.x >> 5);   // wid = m*128+i
  if (wid >= rowsTotal) return;
  int lane = threadIdx.x & 31;
  int m = wid >> 7, i = wid & 127;
  float esv = es[wid];
  float lg[4]; float mx = -__builtin_inff();
#pragma unroll
  for (int t = 0; t < 4; ++t) {
    int j = lane + 32 * t;
    float a = adj[i * Nn + j];
    float v;
    if (a != 0.0f) { float x = esv + ed[m * Nn + j]; v = x > 0.f ? x : 0.2f * x; }
    else v = -__builtin_inff();
    lg[t] = v; mx = fmaxf(mx, v);
  }
  mx = wmaxr(mx);
  float e[4]; float sum = 0.f;
#pragma unroll
  for (int t = 0; t < 4; ++t) {
    e[t] = (lg[t] == -__builtin_inff()) ? 0.f : __expf(lg[t] - mx);
    sum += e[t];
  }
  sum = wsum(sum);
  float inv = sum > 0.f ? 1.0f / sum : 0.0f;   // nan_to_num
#pragma unroll
  for (int t = 0; t < 4; ++t)
    attn[(long long)wid * Nn + lane + 32 * t] = e[t] * inv;
}

// LayerNorm over 256-wide rows: out = (x(+res) - mean)*rsqrt(var+eps)*g + b
template <bool RES, typename OT>
__global__ void k_ln_rows(const float* __restrict__ x, const float* __restrict__ res,
    OT* __restrict__ out, const float* __restrict__ g, const float* __restrict__ b,
    int rows)
{
  int wid = blockIdx.x * 8 + (threadIdx.x >> 5);
  if (wid >= rows) return;
  int lane = threadIdx.x & 31;
  const float* xr = x + (long long)wid * H;
  float v[8]; float s = 0.f;
#pragma unroll
  for (int i = 0; i < 8; ++i) {
    int h = lane + 32 * i;
    float t = xr[h];
    if (RES) t += res[(long long)wid * H + h];
    v[i] = t; s += t;
  }
  s = wsum(s);
  float mean = s * (1.0f / H);
  float q = 0.f;
#pragma unroll
  for (int i = 0; i < 8; ++i) { float d = v[i] - mean; q += d * d; }
  q = wsum(q);
  float inv = rsqrtf(q * (1.0f / H) + EPSF);
#pragma unroll
  for (int i = 0; i < 8; ++i) {
    int h = lane + 32 * i;
    stC(&out[(long long)wid * H + h], (v[i] - mean) * inv * g[h] + b[h]);
  }
}

// seq[.,3t+1,.] = sg (GAT output), sg row = (b*T+t)*N+n
__global__ void k_stack_state(const float* __restrict__ sg, float* __restrict__ seq) {
  int idx = blockIdx.x * 256 + threadIdx.x;
  int h = idx & (H - 1); int r = idx >> 8;
  int n = r % Nn; int m = r / Nn; int b = m / Tt; int t = m % Tt;
  seq[(((long long)(b * Nn + n)) * S3 + 3 * t + 1) * H + h] = sg[idx];
}

// sh gather: sg row = sample*T + t  <-  seq row = sample*60 + 3t+1
__global__ void k_unstack_state(const float* __restrict__ seq, float* __restrict__ sg) {
  int idx = blockIdx.x * 256 + threadIdx.x;
  int h = idx & (H - 1); int r = idx >> 8;
  int t = r % Tt; int smp = r / Tt;
  sg[idx] = seq[(((long long)smp) * S3 + 3 * t + 1) * H + h];
}

// causal attention, one wave per (sample, head, query); S=60, dh=32
__global__ __launch_bounds__(256) void k_attn(const __bf16* __restrict__ qkv,
    float* __restrict__ o, int nSamples)
{
  int wid = blockIdx.x * 8 + (threadIdx.x >> 5);
  int total = nSamples * NH * S3;
  if (wid >= total) return;
  int lane = threadIdx.x & 31;
  int sample = wid / (NH * S3);
  int rem = wid - sample * (NH * S3);
  int head = rem / S3;
  int q = rem - head * S3;

  const __bf16* base = qkv + (long long)sample * S3 * (3 * H);
  const __bf16* qp = base + q * (3 * H) + head * DH;
  float qv[DH];
#pragma unroll
  for (int d = 0; d < DH; ++d) qv[d] = bf2f(qp[d]);

  const float scale = 0.17677669529663687f;   // 1/sqrt(32)
  float s0 = -__builtin_inff(), s1 = -__builtin_inff();
  int j0 = lane, j1 = lane + 32;
  if (j0 <= q) {
    const __bf16* kp = base + j0 * (3 * H) + H + head * DH;
    float s = 0.f;
#pragma unroll
    for (int d = 0; d < DH; ++d) s += qv[d] * bf2f(kp[d]);
    s0 = s * scale;
  }
  if (j1 <= q && j1 < S3) {
    const __bf16* kp = base + j1 * (3 * H) + H + head * DH;
    float s = 0.f;
#pragma unroll
    for (int d = 0; d < DH; ++d) s += qv[d] * bf2f(kp[d]);
    s1 = s * scale;
  }
  float mx = wmaxr(fmaxf(s0, s1));
  float e0 = (s0 == -__builtin_inff()) ? 0.f : __expf(s0 - mx);
  float e1 = (s1 == -__builtin_inff()) ? 0.f : __expf(s1 - mx);
  float sum = wsum(e0 + e1);
  float inv = 1.0f / sum;
  float p0 = e0 * inv, p1 = e1 * inv;

  // lane owns output dim d = lane
  float acc = 0.f;
  for (int jj = 0; jj <= q; ++jj) {
    float p = __shfl(jj < 32 ? p0 : p1, jj & 31, 32);
    const __bf16* vp = base + jj * (3 * H) + 2 * H + head * DH;
    acc += p * bf2f(vp[lane]);
  }
  o[((long long)sample * S3 + q) * H + head * DH + lane] = acc;
}

// output head: logits[r, 0..7] = t[r,:]@W2 + b2
__global__ void k_head(const float* __restrict__ t, const float* __restrict__ W2,
                       const float* __restrict__ b2, float* __restrict__ out, int rows)
{
  int wid = blockIdx.x * 8 + (threadIdx.x >> 5);
  if (wid >= rows) return;
  int lane = threadIdx.x & 31;
  float acc[AD];
#pragma unroll
  for (int j = 0; j < AD; ++j) acc[j] = 0.f;
#pragma unroll
  for (int i = 0; i < 8; ++i) {
    int h = lane + 32 * i;
    float tv = t[(long long)wid * H + h];
#pragma unroll
    for (int j = 0; j < AD; ++j) acc[j] += tv * W2[h * AD + j];
  }
#pragma unroll
  for (int j = 0; j < AD; ++j) {
    float s = wsum(acc[j]);
    if (lane == 0) out[(long long)wid * AD + j] = s + b2[j];
  }
}

// ---------------- host orchestration ----------------
extern "C" void kernel_launch(void* const* d_in, const int* in_sizes, int n_in,
                              void* d_out, int out_size, void* d_ws, size_t ws_size,
                              hipStream_t stream)
{
  (void)in_sizes; (void)n_in; (void)out_size; (void)ws_size;
  auto F = [&](int i) { return (const float*)d_in[i]; };
  const float* states  = F(0);
  const int*   actions = (const int*)d_in[1];
  const float* rtg     = F(2);
  const int*   tsteps  = (const int*)d_in[3];
  const float* adj     = F(4);
  const float *Ws = F(5), *bs = F(6), *Eact = F(7), *Wr = F(8), *br = F(9);
  const float *Et = F(10), *Eag = F(11), *lng = F(12), *lnb = F(13);
  const float *paW1 = F(64), *pab1 = F(65), *paW2 = F(66), *pab2 = F(67);
  float* out = (float*)d_out;

  unsigned char* w = (unsigned char*)d_ws;
  float*  seq   = (float*)w;  w += (size_t)SEQR * H * 4;
  __bf16* hbuf  = (__bf16*)w; w += (size_t)SEQR * H * 2;
  __bf16* chunk = (__bf16*)w; w += (size_t)CH_ROWS * 1024 * 2;
  float*  sg    = (float*)w;  w += (size_t)TOK * H * 4;
  float*  whk   = (float*)w;  w += (size_t)TOK * H * 4;
  float*  outk  = (float*)w;  w += (size_t)TOK * H * 4;
  float*  gacc  = (float*)w;  w += (size_t)TOK * H * 4;
  float*  attnb = (float*)w;  w += (size_t)640 * Nn * Nn * 4;
  float*  es    = (float*)w;  w += (size_t)TOK * 4;
  float*  ed    = (float*)w;  w += (size_t)TOK * 4;

  dim3 blk(256);
  const int EW  = TOK;          // elementwise blocks over TOK*H
  const int WR8 = TOK / 8;      // wave-per-row blocks over TOK rows

  // ---- embeddings ----
  {
    dim3 g(H / BN, TOK / BM, 1);
    gemm_wmma<float, float, 0, false><<<g, blk, 0, stream>>>(
        states, 0, SD, Ws, 0, H, outk, 0, H, SD, nullptr);
  }
  k_embed_state_finish<<<EW, blk, 0, stream>>>(outk, bs, Et, Eag, tsteps, sg);
  k_embed_ret_act<<<EW, blk, 0, stream>>>(rtg, actions, tsteps, Wr, br, Eact, Et, Eag, seq);

  // ---- GAT layers ----
  for (int l = 0; l < GL; ++l) {
    int pb = 14 + 7 * l;
    const float *Wg = F(pb), *asrc = F(pb+1), *adst = F(pb+2), *Wo = F(pb+3);
    const float *bo = F(pb+4), *gg = F(pb+5), *gb = F(pb+6);
    k_fill_bias<<<EW, blk, 0, stream>>>(gacc, bo);
    for (int k = 0; k < GH; ++k) {
      dim3 g1(H / BN, TOK / BM, 1);
      gemm_wmma<float, float, 0, false><<<g1, blk, 0, stream>>>(
          sg, 0, H, Wg + (long long)k * H * H, 0, H, whk, 0, H, H, nullptr);
      k_row_dots<<<WR8, blk, 0, stream>>>(whk, asrc + k * H, adst + k * H, es, ed, TOK);
      k_gat_softmax<<<WR8, blk, 0, stream>>>(es, ed, adj, attnb, TOK);
      dim3 g2(H / BN, 1, 640);   // batched attn(128x128) @ Wh(128x256)
      gemm_wmma<float, float, 0, false><<<g2, blk, 0, stream>>>(
          attnb, (long long)Nn * Nn, Nn, whk, (long long)Nn * H, H,
          outk, (long long)Nn * H, H, Nn, nullptr);
      dim3 g3(H / BN, TOK / BM, 1);
      gemm_wmma<float, float, 0, true><<<g3, blk, 0, stream>>>(
          outk, 0, H, Wo + (long long)k * H * H, 0, H, gacc, 0, H, H, nullptr);
    }
    k_ln_rows<true, float><<<WR8, blk, 0, stream>>>(gacc, sg, sg, gg, gb, TOK);
  }

  // ---- stack + input LN ----
  k_stack_state<<<EW, blk, 0, stream>>>(sg, seq);
  k_ln_rows<false, float><<<SEQR / 8, blk, 0, stream>>>(seq, nullptr, seq, lng, lnb, SEQR);

  // ---- transformer blocks ----
  for (int bI = 0; bI < NL; ++bI) {
    int pb = 28 + 12 * bI;
    const float *g1p = F(pb), *b1p = F(pb+1), *Wqkv = F(pb+2), *bqkv = F(pb+3);
    const float *Wo = F(pb+4), *bo = F(pb+5), *g2p = F(pb+6), *b2p = F(pb+7);
    const float *W1 = F(pb+8), *b1f = F(pb+9), *W2 = F(pb+10), *b2f = F(pb+11);

    k_ln_rows<false, __bf16><<<SEQR / 8, blk, 0, stream>>>(seq, nullptr, hbuf, g1p, b1p, SEQR);
    for (int c = 0; c < 4; ++c) {
      long long ro = (long long)c * CH_ROWS;
      { dim3 g(768 / BN, CH_ROWS / BM, 1);
        gemm_wmma<__bf16, __bf16, 1, false><<<g, blk, 0, stream>>>(
            hbuf + ro * H, 0, H, Wqkv, 0, 768, chunk, 0, 768, H, bqkv); }
      k_attn<<<CH_ROWS, blk, 0, stream>>>(chunk, outk, CH_SAMP);
      { dim3 g(H / BN, CH_ROWS / BM, 1);
        gemm_wmma<float, float, 1, true><<<g, blk, 0, stream>>>(
            outk, 0, H, Wo, 0, H, seq + ro * H, 0, H, H, bo); }
    }
    k_ln_rows<false, __bf16><<<SEQR / 8, blk, 0, stream>>>(seq, nullptr, hbuf, g2p, b2p, SEQR);
    for (int c = 0; c < 4; ++c) {
      long long ro = (long long)c * CH_ROWS;
      { dim3 g(1024 / BN, CH_ROWS / BM, 1);
        gemm_wmma<__bf16, __bf16, 2, false><<<g, blk, 0, stream>>>(
            hbuf + ro * H, 0, H, W1, 0, 1024, chunk, 0, 1024, H, b1f); }
      { dim3 g(H / BN, CH_ROWS / BM, 1);
        gemm_wmma<__bf16, float, 1, true><<<g, blk, 0, stream>>>(
            chunk, 0, 1024, W2, 0, H, seq + ro * H, 0, H, 1024, b2f); }
    }
  }

  // ---- output head ----
  k_unstack_state<<<EW, blk, 0, stream>>>(seq, sg);
  { dim3 g(H / BN, TOK / BM, 1);
    gemm_wmma<float, float, 2, false><<<g, blk, 0, stream>>>(
        sg, 0, H, paW1, 0, H, whk, 0, H, H, pab1); }
  k_head<<<WR8, blk, 0, stream>>>(whk, paW2, pab2, out, TOK);
}
